// GraphSAGE_86242943303862
// MI455X (gfx1250) — matrix operand order repaired and checked by reading, not compiled, not run on hardware
//
#include <hip/hip_runtime.h>
#include <hip/hip_bf16.h>

#define D 128

typedef __attribute__((ext_vector_type(16))) _Float16 v16h;
typedef __attribute__((ext_vector_type(4)))  _Float16 v4h;
typedef __attribute__((ext_vector_type(8)))  float    v8f;

union AU { v16h v; unsigned u[8]; };
union BU { v16h v; uint4 q[2]; };
union PK { _Float16 h[2]; unsigned u; };

// ---------------- degree / invdeg ----------------
__global__ __launch_bounds__(256) void deg_kernel(const int* __restrict__ dst,
                                                  float* __restrict__ deg, int E) {
  int e = blockIdx.x * 256 + threadIdx.x;
  if (e < E) atomicAdd(&deg[dst[e]], 1.0f);
}

__global__ __launch_bounds__(256) void invdeg_kernel(float* deg, int N) {
  int i = blockIdx.x * 256 + threadIdx.x;
  if (i < N) deg[i] = 1.0f / fmaxf(deg[i], 1.0f);
}

// ---------------- f32 -> f16 convert ----------------
__global__ __launch_bounds__(256) void cvt_f16(const float* __restrict__ src,
                                               _Float16* __restrict__ dst, long long n) {
  long long i = (long long)blockIdx.x * 256 + threadIdx.x;
  if (i < n) dst[i] = (_Float16)src[i];
}

// ---------------- mean-scale + f16 convert ----------------
__global__ __launch_bounds__(256) void scale_cvt(const float* __restrict__ agg,
                                                 const float* __restrict__ invdeg,
                                                 _Float16* __restrict__ out, int N) {
  long long i = (long long)blockIdx.x * 256 + threadIdx.x;
  long long n = (long long)N * D;
  if (i < n) out[i] = (_Float16)(agg[i] * invdeg[i >> 7]);
}

// ---------------- pack weights into WMMA B-operand layout ----------------
// B[k][n] = Wcat[n][k], Wcat = [W0 ; W1] (K rows). Layout per (ktile t, ntile j):
// 32 lanes x 8 dwords; lane L: n = 16j + (L&15), kbase = 16*(L>>4); dword v holds
// halfs (kbase+2v, kbase+2v+1).  Flat dword index = ((t*8+j)*32+lane)*8+v = tid.
__global__ __launch_bounds__(256) void pack_w(const float* __restrict__ W0,
                                              const float* __restrict__ W1,
                                              unsigned* __restrict__ dst, int K) {
  int tid = blockIdx.x * 256 + threadIdx.x;
  if (tid >= K * 64) return;
  int v = tid & 7, lane = (tid >> 3) & 31, j = (tid >> 8) & 7, t = tid >> 11;
  int n = 16 * j + (lane & 15);
  int k = 32 * t + 16 * (lane >> 4) + 2 * v;
  const float* W = (k < D) ? W0 : W1;
  int kk = (k < D) ? k : (k - D);
  PK p;
  p.h[0] = (_Float16)W[n * D + kk];
  p.h[1] = (_Float16)W[n * D + kk + 1];
  dst[tid] = p.u;
}

// ---------------- edge gather: agg[dst] += f16 X[src] ----------------
__global__ __launch_bounds__(256) void gather_kernel(const _Float16* __restrict__ X,
                                                     const int* __restrict__ srcI,
                                                     const int* __restrict__ dstI,
                                                     float* __restrict__ agg, int E) {
  int e = blockIdx.x * 8 + (threadIdx.x >> 5);  // one wave32 per edge
  if (e >= E) return;
  int lane = threadIdx.x & 31;
  int c = lane * 4;
  const v4h hv = *(const v4h*)(X + (long long)srcI[e] * D + c);
  float* ap = agg + (long long)dstI[e] * D + c;
  atomicAdd(ap + 0, (float)hv.x);
  atomicAdd(ap + 1, (float)hv.y);
  atomicAdd(ap + 2, (float)hv.z);
  atomicAdd(ap + 3, (float)hv.w);
}

// ---------------- WMMA GEMM: out = act([A0|A1] @ packedB + bias) ----------------
// Each wave: 16 rows x 128 cols; K in {128, 256}. B staged in LDS (K*256 bytes).
__global__ __launch_bounds__(256) void gemm_wmma(
    const _Float16* __restrict__ A0, const _Float16* __restrict__ A1,
    const unsigned* __restrict__ Bp, const float* __restrict__ bias,
    float* __restrict__ outF, _Float16* __restrict__ outH,
    int nrows, int K, int relu) {
  extern __shared__ unsigned lb[];
  const int tid = threadIdx.x;

  // stage packed B into LDS (uint4 = ds_store_b128)
  const int nq = (K * 64) >> 2;
  const uint4* gq = (const uint4*)Bp;
  uint4* lq = (uint4*)lb;
  for (int i = tid; i < nq; i += 256) lq[i] = gq[i];
  __syncthreads();

  const int lane = tid & 31;
  const int wave = tid >> 5;
  const int m = lane & 15;
  const int hh = lane >> 4;
  const int row0 = blockIdx.x * 128 + wave * 16;
  int row = row0 + m;
  if (row >= nrows) row = nrows - 1;  // clamp: keep EXEC all-ones for WMMA
  const long long rb = (long long)row * D;

  v8f z = {};
  v8f acc[8];
#pragma unroll
  for (int j = 0; j < 8; ++j) acc[j] = z;

  const int kt = K >> 5;
  for (int t = 0; t < kt; ++t) {
    AU a;
#pragma unroll
    for (int v = 0; v < 8; ++v) {
      // A k order used here (same as packed B): VGPR v -> k = 2v + 16*(lane>>4)
      int k = 2 * v + 16 * hh;
      int kg = (t << 5) + k;
      const _Float16* Ap = (kg < D) ? (A0 + rb + kg) : (A1 + rb + (kg - D));
      a.u[v] = *(const unsigned*)Ap;
    }
#pragma unroll
    for (int j = 0; j < 8; ++j) {
      BU b;
      const uint4* lp = (const uint4*)&lb[((t * 8 + j) * 32 + lane) * 8];
      b.q[0] = lp[0];
      b.q[1] = lp[1];
      acc[j] = __builtin_amdgcn_wmma_f32_16x16x32_f16(
          false, a.v, false, b.v, (short)0, acc[j], false, false);
    }
  }

#pragma unroll
  for (int j = 0; j < 8; ++j) {
    const int col = 16 * j + m;
    const float bc = bias[col];
#pragma unroll
    for (int r = 0; r < 8; ++r) {
      int rr = row0 + 8 * hh + r;  // C layout: VGPR r: M = r (lanes 0-15), r+8 (16-31)
      float val = acc[j][r] + bc;
      if (relu) val = fmaxf(val, 0.0f);
      if (rr < nrows) {
        long long o = (long long)rr * D + col;
        if (outF) outF[o] = val;
        if (outH) outH[o] = (_Float16)val;
      }
    }
  }
}

extern "C" void kernel_launch(void* const* d_in, const int* in_sizes, int n_in,
                              void* d_out, int out_size, void* d_ws, size_t ws_size,
                              hipStream_t stream) {
  const float* x   = (const float*)d_in[0];
  const int*   ei  = (const int*)d_in[1];
  const float* Wl0 = (const float*)d_in[2];
  const float* bl0 = (const float*)d_in[3];
  const float* Wr0 = (const float*)d_in[4];
  const float* Wl1 = (const float*)d_in[5];
  const float* bl1 = (const float*)d_in[6];
  const float* Wr1 = (const float*)d_in[7];
  const float* Wv  = (const float*)d_in[8];
  const float* bv  = (const float*)d_in[9];
  const float* Wt  = (const float*)d_in[10];
  const float* bt  = (const float*)d_in[11];

  const int N = in_sizes[0] / D;
  const int E = in_sizes[1] / 2;
  const int* srcI = ei;
  const int* dstI = ei + E;

  char* ws = (char*)d_ws;
  size_t off = 0;
  auto take = [&](size_t bytes) -> void* {
    void* p = ws + off;
    off += (bytes + 255) & ~(size_t)255;
    return p;
  };
  float*    deg  = (float*)take((size_t)N * 4);
  _Float16* xh   = (_Float16*)take((size_t)N * D * 2);   // f16(x); later reused as f16(h)
  _Float16* h0h  = (_Float16*)take((size_t)N * D * 2);   // f16(layer0 output)
  _Float16* aggh = (_Float16*)take((size_t)N * D * 2);   // f16(mean aggregation)
  unsigned* w0p  = (unsigned*)take((size_t)2 * D * D * 2);
  unsigned* w1p  = (unsigned*)take((size_t)2 * D * D * 2);
  unsigned* wvp  = (unsigned*)take((size_t)D * D * 2);
  unsigned* wtp  = (unsigned*)take((size_t)D * D * 2);

  float* outHh = (float*)d_out;                    // h
  float* outV  = (float*)d_out + (size_t)N * D;    // x_vision
  float* outT  = (float*)d_out + (size_t)2 * N * D;// x_text
  float* agg   = outV;  // f32 scatter buffer aliases x_vision slot (overwritten later)

  const long long nd = (long long)N * D;
  const int eltBlocks = (int)((nd + 255) / 256);
  const int gemmBlocks = (N + 127) / 128;

  hipMemsetAsync(deg, 0, (size_t)N * 4, stream);
  hipMemsetAsync(agg, 0, (size_t)N * D * 4, stream);

  deg_kernel<<<(E + 255) / 256, 256, 0, stream>>>(dstI, deg, E);
  invdeg_kernel<<<(N + 255) / 256, 256, 0, stream>>>(deg, N);
  cvt_f16<<<eltBlocks, 256, 0, stream>>>(x, xh, nd);

  pack_w<<<(2 * D * 64) / 256, 256, 0, stream>>>(Wl0, Wr0, w0p, 2 * D);
  pack_w<<<(2 * D * 64) / 256, 256, 0, stream>>>(Wl1, Wr1, w1p, 2 * D);
  pack_w<<<(D * 64) / 256, 256, 0, stream>>>(Wv, nullptr, wvp, D);
  pack_w<<<(D * 64) / 256, 256, 0, stream>>>(Wt, nullptr, wtp, D);

  // ---- layer 0 ----
  gather_kernel<<<(E + 7) / 8, 256, 0, stream>>>(xh, srcI, dstI, agg, E);
  scale_cvt<<<eltBlocks, 256, 0, stream>>>(agg, deg, aggh, N);
  // h0 = relu([agg|x] @ [Wl0;Wr0]^T + bl0), stored f16
  gemm_wmma<<<gemmBlocks, 256, 2 * D * 256, stream>>>(aggh, xh, w0p, bl0,
                                                      nullptr, h0h, N, 2 * D, 1);

  // ---- layer 1 ----
  hipMemsetAsync(agg, 0, (size_t)N * D * 4, stream);
  gather_kernel<<<(E + 7) / 8, 256, 0, stream>>>(h0h, srcI, dstI, agg, E);
  scale_cvt<<<eltBlocks, 256, 0, stream>>>(agg, deg, aggh, N);
  // h = [agg|h0] @ [Wl1;Wr1]^T + bl1 -> f32 out + f16 copy (reuse xh)
  gemm_wmma<<<gemmBlocks, 256, 2 * D * 256, stream>>>(aggh, h0h, w1p, bl1,
                                                      outHh, xh, N, 2 * D, 0);

  // ---- heads ----
  gemm_wmma<<<gemmBlocks, 256, D * 256, stream>>>(xh, nullptr, wvp, bv,
                                                  outV, nullptr, N, D, 1);
  gemm_wmma<<<gemmBlocks, 256, D * 256, stream>>>(xh, nullptr, wtp, bt,
                                                  outT, nullptr, N, D, 1);
}
